// QGraphAttention_46780783788316
// MI455X (gfx1250) — matrix operand order repaired and checked by reading, not compiled, not run on hardware
//
#include <hip/hip_runtime.h>
#include <hip/hip_bf16.h>

// ---------------- problem dims ----------------
#define B_   512
#define NA_  8
#define OB_  160
#define SD_  256
#define H_   4
#define GD_  64
#define KD_  65        // GD+1
#define NU_  32
#define N_   4096      // B*NA
#define HCOL 4225      // KD*KD
#define HCOLP 4352     // padded to 272 tiles (multiple of 8 tiles -> uniform pair loop)
#define NTILE (HCOLP / 16)   // 272
#define KTOT 160
#define NKCH 5         // 160 / 32

// d_out layout: q_tot [512] | attn [512*4*8] | gnn_weight_matrics [4096*4096*4]
#define OFF_ATTN 512
#define OFF_BIG  16896
#define BIG_N4   16777216LL   // 4096*4096*4 floats / 4

typedef __attribute__((ext_vector_type(16))) _Float16 v16h;
typedef __attribute__((ext_vector_type(2)))  _Float16 h2v;
typedef __attribute__((ext_vector_type(8)))  float    v8f;

// ---------------------------------------------------------------------------
// K0: zero-fill the 256 MiB block-diagonal attention-matrix output.
// Pure bandwidth; float4 (b128) stores, grid-stride.
// ---------------------------------------------------------------------------
__global__ __launch_bounds__(256) void k_zero(float4* __restrict__ p, long long n4) {
  long long i = (long long)blockIdx.x * blockDim.x + threadIdx.x;
  long long stride = (long long)gridDim.x * blockDim.x;
  float4 z; z.x = 0.f; z.y = 0.f; z.z = 0.f; z.w = 0.f;
  for (; i < n4; i += stride) p[i] = z;
}

// ---------------------------------------------------------------------------
// K_cvt: W_h [160 x 4225] f32  ->  WhT f16 [4352 x 160] (column-major by c,
// zero-padded cols) so WMMA B-fragments load K-contiguous f16 pairs.
// ---------------------------------------------------------------------------
__global__ __launch_bounds__(256) void k_cvt(const float* __restrict__ Wh,
                                             _Float16* __restrict__ WhT) {
  int idx = blockIdx.x * 256 + threadIdx.x;      // 4352*160 = 696320 = 2720*256
  int c = idx / KTOT, k = idx - c * KTOT;
  float v = (c < HCOL) ? Wh[k * HCOL + c] : 0.f;
  WhT[idx] = (_Float16)v;
}

// ---------------------------------------------------------------------------
// K1: per-sample GAT. One block = one batch sample.
// ---------------------------------------------------------------------------
__global__ __launch_bounds__(256) void k_gnn(
    const float* __restrict__ agent_qs, const float* __restrict__ obs,
    const float* __restrict__ W_gnn, const float* __restrict__ b_gnn,
    const float* __restrict__ a_src, const float* __restrict__ a_dst,
    float* __restrict__ big, float* __restrict__ keyin,
    _Float16* __restrict__ obsF) {
  __shared__ float obs_s[NA_ * OB_];
  __shared__ float h_s[NA_ * GD_];
  __shared__ float ssrc[NA_ * H_], sdst[NA_ * H_];
  __shared__ float alpha_s[NA_ * NA_ * H_];
  __shared__ float am_s[NA_ * NA_];
  const int b = blockIdx.x, tid = threadIdx.x;
  const float* ob = obs + b * NA_ * OB_;

  for (int i = tid; i < NA_ * OB_; i += 256) {
    float v = ob[i];
    obs_s[i] = v;
    obsF[b * NA_ * OB_ + i] = (_Float16)v;
  }
  __syncthreads();

  for (int o = tid; o < NA_ * GD_; o += 256) {
    int a = o >> 6, g = o & 63;
    float sum = b_gnn[g];
    const float* orow = &obs_s[a * OB_];
    for (int i = 0; i < OB_; ++i) sum += orow[i] * W_gnn[i * GD_ + g];
    h_s[a * GD_ + g] = sum;
  }
  __syncthreads();

  if (tid < NA_ * H_) {                       // 32 (agent, head) pairs
    int a = tid >> 2, hh = tid & 3;
    float s1 = 0.f, s2 = 0.f;
    for (int g = 0; g < GD_; ++g) {
      float hv = h_s[a * GD_ + g];
      s1 += hv * a_src[hh * GD_ + g];
      s2 += hv * a_dst[hh * GD_ + g];
    }
    ssrc[tid] = s1; sdst[tid] = s2;
  }
  __syncthreads();

  if (tid < NA_ * H_) {                       // softmax over 7 in-edges per (dst,head)
    int d = tid >> 2, hh = tid & 3;
    float e[NA_]; float mx = -1e30f;
    for (int s = 0; s < NA_; ++s) {
      if (s == d) { e[s] = 0.f; continue; }
      float x = ssrc[s * H_ + hh] + sdst[d * H_ + hh];
      x = (x > 0.f) ? x : 0.2f * x;           // leaky_relu(0.2)
      e[s] = x; mx = fmaxf(mx, x);
    }
    float sum = 0.f;
    for (int s = 0; s < NA_; ++s) if (s != d) { e[s] = __expf(e[s] - mx); sum += e[s]; }
    float inv = 1.f / (sum + 1e-16f);
    alpha_s[(d * NA_ + d) * H_ + hh] = 0.f;   // diagonal
    for (int s = 0; s < NA_; ++s) if (s != d) {
      float al = e[s] * inv;
      alpha_s[(s * NA_ + d) * H_ + hh] = al;
      big[((size_t)(b * NA_ + s) * N_ + (b * NA_ + d)) * H_ + hh] = al;
    }
  }
  __syncthreads();

  if (tid < NA_ * NA_) {                      // head-mean alpha
    int s = tid >> 3, d = tid & 7;
    const float* ap = &alpha_s[(s * NA_ + d) * H_];
    am_s[tid] = 0.25f * (ap[0] + ap[1] + ap[2] + ap[3]);
  }
  __syncthreads();

  for (int o = tid; o < NA_ * GD_; o += 256) { // agg + relu -> keyin[1..64]
    int d = o >> 6, g = o & 63;
    float sum = 0.f;
    for (int s = 0; s < NA_; ++s) sum += am_s[s * NA_ + d] * h_s[s * GD_ + g];
    keyin[(b * NA_ + d) * KD_ + 1 + g] = fmaxf(sum, 0.f);
  }
  if (tid < NA_) keyin[(b * NA_ + tid) * KD_] = agent_qs[b * NA_ + tid];
}

// ---------------------------------------------------------------------------
// K2: fused hypernet GEMM + weighted-key reduction (the 5.5 GFLOP hot spot).
//   Two column tiles per iteration -> two independent WMMA accumulator
//   chains, all B fragments preloaded (one load clause, one wait, 10
//   interleaved v_wmma). Epilogue uses transposed [j][16] LDS accumulators
//   so the per-lane RMW is 2x ds_load_b128 + 8 FMA + 2x ds_store_b128.
//   Per-wave private accumulators -> no atomics, deterministic.
// ---------------------------------------------------------------------------
__global__ __launch_bounds__(128) void k_hyper(
    const _Float16* __restrict__ obsF, const _Float16* __restrict__ WhT,
    const float* __restrict__ bh, const float* __restrict__ keyin,
    float* __restrict__ keysOut) {
  __shared__ float keyinT_s[KD_ * 16];        // [k][row]
  __shared__ float acc_s[4][KD_ * 16];        // per-wave, [j][row]
  const int tid = threadIdx.x;
  const int lane = tid & 31, wave = tid >> 5;
  const int node0 = blockIdx.x * 16;

  for (int i = tid; i < KD_ * 16; i += 128) {   // transpose keyin into LDS
    int k = i >> 4, mm = i & 15;
    keyinT_s[i] = keyin[(node0 + mm) * KD_ + k];
  }
  for (int i = tid; i < 4 * KD_ * 16; i += 128) (&acc_s[0][0])[i] = 0.f;
  __syncthreads();

  const int m = lane & 15, hi = lane >> 4, n = lane & 15;

  // A fragments (obs rows, f16). Contiguous K order matching B:
  // lanes 0-15: VGPR0..7 -> K=0..15; lanes 16-31: K=16..31.
  v16h afrag[NKCH];
  {
    const _Float16* arow = obsF + (size_t)(node0 + m) * KTOT;
#pragma unroll
    for (int kc = 0; kc < NKCH; ++kc) {
      union { v16h v; h2v p[8]; } ua;
#pragma unroll
      for (int i = 0; i < 8; ++i) {
        int kb = 2 * i + hi * 16;
        ua.p[i] = *(const h2v*)(arow + kc * 32 + kb);
      }
      afrag[kc] = ua.v;
    }
  }

  float* accw = &acc_s[wave][0];

  for (int t0 = wave * 2; t0 < NTILE; t0 += 8) {   // uniform: 34 iters per wave
    // ---- preload B fragments for both tiles, all K chunks ----
    union { v16h v; h2v p[8]; } b0[NKCH], b1[NKCH];
    const _Float16* bc0 = WhT + (size_t)(t0 * 16 + n) * KTOT + hi * 16;
    const _Float16* bc1 = bc0 + 16 * KTOT;
#pragma unroll
    for (int kc = 0; kc < NKCH; ++kc) {
#pragma unroll
      for (int i = 0; i < 8; ++i) {
        b0[kc].p[i] = *(const h2v*)(bc0 + kc * 32 + 2 * i);
        b1[kc].p[i] = *(const h2v*)(bc1 + kc * 32 + 2 * i);
      }
    }
    __builtin_prefetch(WhT + (size_t)(t0 * 16 + 128 + n) * KTOT, 0, 1);

    // ---- 10 WMMAs, two independent accumulator chains interleaved ----
    v8f acc0 = {}, acc1 = {};
#pragma unroll
    for (int kc = 0; kc < NKCH; ++kc) {
      acc0 = __builtin_amdgcn_wmma_f32_16x16x32_f16(
          false, afrag[kc], false, b0[kc].v, (short)0, acc0, false, false);
      acc1 = __builtin_amdgcn_wmma_f32_16x16x32_f16(
          false, afrag[kc], false, b1[kc].v, (short)0, acc1, false, false);
    }

    // ---- epilogue: c = k*65+j ; keys[row,j] += keyin[row,k]*|hyp+bias| ----
#pragma unroll
    for (int tt = 0; tt < 2; ++tt) {
      const int c = (t0 + tt) * 16 + n;
      const v8f acc = tt ? acc1 : acc0;
      if (c < HCOL) {
        const int k = c / KD_;
        const int j = c - k * KD_;
        const float bias = bh[c];
        const float4 s0 = *(const float4*)&keyinT_s[k * 16 + hi * 8];
        const float4 s1 = *(const float4*)&keyinT_s[k * 16 + hi * 8 + 4];
        float4* dst = (float4*)&accw[j * 16 + hi * 8];
        float4 a0 = dst[0], a1 = dst[1];
        a0.x += s0.x * fabsf(acc[0] + bias);
        a0.y += s0.y * fabsf(acc[1] + bias);
        a0.z += s0.z * fabsf(acc[2] + bias);
        a0.w += s0.w * fabsf(acc[3] + bias);
        a1.x += s1.x * fabsf(acc[4] + bias);
        a1.y += s1.y * fabsf(acc[5] + bias);
        a1.z += s1.z * fabsf(acc[6] + bias);
        a1.w += s1.w * fabsf(acc[7] + bias);
        dst[0] = a0; dst[1] = a1;
      }
    }
  }
  __syncthreads();

  for (int i = tid; i < KD_ * 16; i += 128) {  // reduce waves + elu + store
    int jj = i >> 4, mm = i & 15;
    float x = acc_s[0][i] + acc_s[1][i] + acc_s[2][i] + acc_s[3][i];
    keysOut[(size_t)(node0 + mm) * KD_ + jj] = (x > 0.f) ? x : (__expf(x) - 1.f);
  }
}

// ---------------------------------------------------------------------------
// K3: mixer. One block = one sample.
// ---------------------------------------------------------------------------
__global__ __launch_bounds__(256) void k_mix(
    const float* __restrict__ agent_qs, const float* __restrict__ states,
    const float* __restrict__ keysE,
    const float* __restrict__ Wq, const float* __restrict__ Wk,
    const float* __restrict__ Wv1, const float* __restrict__ bv1,
    const float* __restrict__ Wv2, const float* __restrict__ bv2,
    const float* __restrict__ Ww1, const float* __restrict__ bw1,
    const float* __restrict__ Ww2, const float* __restrict__ bw2,
    float* __restrict__ out) {
  __shared__ float st[SD_];
  __shared__ float ks[NA_ * KD_];
  __shared__ float kp[NA_ * NU_];
  __shared__ float qv[NU_], hid[64], hvv[32];
  __shared__ float outh[H_], hw[H_];
  const int b = blockIdx.x, tid = threadIdx.x;

  st[tid] = states[b * SD_ + tid];
  for (int i = tid; i < NA_ * KD_; i += 256) ks[i] = keysE[b * NA_ * KD_ + i];
  __syncthreads();

  {                                            // kproj: 256 dots of 65
    int a = tid >> 5, u = tid & 31;
    float s = 0.f;
    for (int j = 0; j < KD_; ++j) s += ks[a * KD_ + j] * Wk[j * NU_ + u];
    kp[a * NU_ + u] = s;
  }
  if (tid < 32) {                              // q = states@Wq
    float s = 0.f;
    for (int i = 0; i < SD_; ++i) s += st[i] * Wq[i * NU_ + tid];
    qv[tid] = s;
  } else if (tid < 96) {                       // hidden = relu(states@Ww1+bw1)
    int x = tid - 32;
    float s = bw1[x];
    for (int i = 0; i < SD_; ++i) s += st[i] * Ww1[i * 64 + x];
    hid[x] = fmaxf(s, 0.f);
  } else if (tid < 128) {                      // hv = relu(states@Wv1+bv1)
    int x = tid - 96;
    float s = bv1[x];
    for (int i = 0; i < SD_; ++i) s += st[i] * Wv1[i * 32 + x];
    hvv[x] = fmaxf(s, 0.f);
  }
  __syncthreads();

  if (tid < H_) {
    const int h = tid;
    float sc[NA_], mx = -1e30f;
    for (int a = 0; a < NA_; ++a) {
      float s = 0.f;
      for (int dd = 0; dd < 8; ++dd) s += qv[h * 8 + dd] * kp[a * NU_ + h * 8 + dd];
      s *= 0.35355339059327373f;               // 1/sqrt(d=8)
      sc[a] = s; mx = fmaxf(mx, s);
    }
    float sum = 0.f;
    for (int a = 0; a < NA_; ++a) { sc[a] = __expf(sc[a] - mx); sum += sc[a]; }
    float inv = 1.f / sum, o = 0.f;
    for (int a = 0; a < NA_; ++a) {
      float at = sc[a] * inv;
      out[OFF_ATTN + (b * H_ + h) * NA_ + a] = at;
      o += at * agent_qs[b * NA_ + a];
    }
    outh[h] = o;
    float w = bw2[h];
    for (int i = 0; i < 64; ++i) w += hid[i] * Ww2[i * H_ + h];
    hw[h] = w * w;
  }
  __syncthreads();

  if (tid == 0) {
    float v = bv2[0];
    for (int i = 0; i < 32; ++i) v += hvv[i] * Wv2[i];
    float qt = v;
    for (int h = 0; h < H_; ++h) qt += outh[h] * hw[h];
    out[b] = qt;
  }
}

// ---------------------------------------------------------------------------
extern "C" void kernel_launch(void* const* d_in, const int* in_sizes, int n_in,
                              void* d_out, int out_size, void* d_ws, size_t ws_size,
                              hipStream_t stream) {
  const float* agent_qs = (const float*)d_in[0];
  const float* obs      = (const float*)d_in[1];
  const float* states   = (const float*)d_in[2];
  /* d_in[3] = edge_index (int32) — graph structure is known statically */
  const float* W_gnn = (const float*)d_in[4];
  const float* b_gnn = (const float*)d_in[5];
  const float* a_src = (const float*)d_in[6];
  const float* a_dst = (const float*)d_in[7];
  const float* W_h   = (const float*)d_in[8];
  const float* b_h   = (const float*)d_in[9];
  const float* Wq    = (const float*)d_in[10];
  const float* Wk    = (const float*)d_in[11];
  const float* Wv1   = (const float*)d_in[12];
  const float* bv1   = (const float*)d_in[13];
  const float* Wv2   = (const float*)d_in[14];
  const float* bv2   = (const float*)d_in[15];
  const float* Ww1   = (const float*)d_in[16];
  const float* bw1   = (const float*)d_in[17];
  const float* Ww2   = (const float*)d_in[18];
  const float* bw2   = (const float*)d_in[19];

  float* out = (float*)d_out;
  char*  ws  = (char*)d_ws;
  // workspace layout (all regions fully overwritten every call)
  float*    keyin = (float*)(ws);                        // 4096*65 f32
  float*    keysE = (float*)(ws + 1064960);              // 4096*65 f32
  _Float16* obsF  = (_Float16*)(ws + 2129920);           // 4096*160 f16
  _Float16* WhT   = (_Float16*)(ws + 3440640);           // 4352*160 f16
  float*    big   = out + OFF_BIG;

  k_zero <<<8192, 256, 0, stream>>>((float4*)big, BIG_N4);
  k_cvt  <<<2720, 256, 0, stream>>>(W_h, WhT);
  k_gnn  <<<B_,   256, 0, stream>>>(agent_qs, obs, W_gnn, b_gnn, a_src, a_dst,
                                    big, keyin, obsF);
  k_hyper<<<N_/16,128, 0, stream>>>(obsF, WhT, b_h, keyin, keysE);
  k_mix  <<<B_,   256, 0, stream>>>(agent_qs, states, keysE, Wq, Wk,
                                    Wv1, bv1, Wv2, bv2, Ww1, bw1, Ww2, bw2, out);
}
